// GNNMambaFusion_15161234555023
// MI455X (gfx1250) — matrix operand order repaired and checked by reading, not compiled, not run on hardware
//
#include <hip/hip_runtime.h>
#include <hip/hip_bf16.h>
#include <math.h>

#define H_DIM   128
#define L_DIM   128
#define G_DIM   512
#define N_NODES (G_DIM * L_DIM)   /* 65536 */
#define E_EDGES 524288
#define NL_LAYERS 4
#define DT_DIM  8
#define NS_DIM  2
#define DC_DIM  4
#define XDBL_DIM (DT_DIM + 2 * NS_DIM) /* 12 */

typedef __attribute__((ext_vector_type(16))) _Float16 v16h;
typedef __attribute__((ext_vector_type(8)))  _Float16 v8h;
typedef __attribute__((ext_vector_type(8)))  float    v8f;

// gfx1250 async copy to LDS (ASYNCcnt path), guarded so the source always compiles
#if defined(__gfx1250__) && __has_builtin(__builtin_amdgcn_global_load_async_to_lds_b128) && __has_builtin(__builtin_amdgcn_s_wait_asynccnt)
#define HAVE_ASYNC_LDS 1
// Builtin prototype (from clang diagnostics): vV4i*1 V4i*3 Ii Ii
// -> GCC-style int4 pointers in addrspace(1) (global) and addrspace(3) (LDS).
typedef int v4i_g __attribute__((vector_size(16)));
typedef __attribute__((address_space(1))) v4i_g* as1_v4p;
typedef __attribute__((address_space(3))) v4i_g* as3_v4p;
#else
#define HAVE_ASYNC_LDS 0
#endif

// ---------------------------------------------------------------------------
// helpers
// ---------------------------------------------------------------------------
__device__ __forceinline__ v16h pack16(float4 u0, float4 u1, float4 u2, float4 u3) {
    v16h a;
    a[0]=(_Float16)u0.x;  a[1]=(_Float16)u0.y;  a[2]=(_Float16)u0.z;  a[3]=(_Float16)u0.w;
    a[4]=(_Float16)u1.x;  a[5]=(_Float16)u1.y;  a[6]=(_Float16)u1.z;  a[7]=(_Float16)u1.w;
    a[8]=(_Float16)u2.x;  a[9]=(_Float16)u2.y;  a[10]=(_Float16)u2.z; a[11]=(_Float16)u2.w;
    a[12]=(_Float16)u3.x; a[13]=(_Float16)u3.y; a[14]=(_Float16)u3.z; a[15]=(_Float16)u3.w;
    return a;
}
__device__ __forceinline__ float4 fuse4(float4 a, float4 b, float e) {
    float4 r; r.x = e*a.x + b.x; r.y = e*a.y + b.y; r.z = e*a.z + b.z; r.w = e*a.w + b.w;
    return r;
}
__device__ __forceinline__ float siluf(float x) { return x / (1.0f + expf(-x)); }

// ---------------------------------------------------------------------------
// Generic K=128 WMMA GEMM:  C[M,NOUT] = act(A' @ W^T + bias)
//   A' = FUSE ? (1+eps)*A + A2 : A        (A, A2 row-major M x 128, f32)
//   W row-major NOUT x 128 f32, staged to LDS as f16 (async-copy staged when
//   the gfx1250 ASYNC builtins are available and NOUT==128).
// Block = 256 threads (8 waves); each wave owns a 16-row M tile; block = 128 rows.
// ---------------------------------------------------------------------------
template<int NOUT, bool FUSE, bool RELU, bool BIAS>
__global__ __launch_bounds__(256)
void gemm_k128_wmma(const float* __restrict__ A,
                    const float* __restrict__ A2,
                    const float* __restrict__ epsp,
                    const float* __restrict__ W,
                    const float* __restrict__ bias,
                    float* __restrict__ C,
                    int ldc)
{
    __shared__ __align__(32) _Float16 wlds[NOUT * 128];
    const int tid = threadIdx.x;

#if HAVE_ASYNC_LDS
    if constexpr (NOUT == 128) {
        // Stage raw f32 weights into LDS with GLOBAL_LOAD_ASYNC_TO_LDS_B128
        // (tracked by ASYNCcnt), then convert to f16 fragments in LDS.
        __shared__ __align__(16) float wstage[32 * 128];   // 16 KB chunk buffer
#pragma unroll 1
        for (int chunk = 0; chunk < 4; ++chunk) {
            const float* gsrc = W + chunk * (32 * 128);
            for (int idx = tid; idx < 32 * 128 / 4; idx += 256) {
                __builtin_amdgcn_global_load_async_to_lds_b128(
                    (as1_v4p)(gsrc + idx * 4),
                    (as3_v4p)(&wstage[idx * 4]),
                    0, 0);
            }
            __builtin_amdgcn_s_wait_asynccnt(0);
            __syncthreads();
            for (int idx = tid; idx < 32 * 128; idx += 256)
                wlds[chunk * (32 * 128) + idx] = (_Float16)wstage[idx];
            __syncthreads();
        }
    } else
#endif
    {
        // direct load + convert path
        for (int idx = tid; idx < NOUT * 128 / 8; idx += 256) {
            const int base = idx * 8;
            const float4* wp = reinterpret_cast<const float4*>(W + base);
            float4 w0 = wp[0], w1 = wp[1];
            _Float16* dst = &wlds[base];
            dst[0]=(_Float16)w0.x; dst[1]=(_Float16)w0.y; dst[2]=(_Float16)w0.z; dst[3]=(_Float16)w0.w;
            dst[4]=(_Float16)w1.x; dst[5]=(_Float16)w1.y; dst[6]=(_Float16)w1.z; dst[7]=(_Float16)w1.w;
        }
        __syncthreads();
    }

    const int lane  = tid & 31;
    const int wave  = tid >> 5;
    const int row16 = lane & 15;
    const int hi    = lane >> 4;       // 0 or 1
    const size_t m_base = (size_t)blockIdx.x * 128 + (size_t)wave * 16;

    const float* Arow  = A + (m_base + row16) * 128;
    const float* A2row = nullptr;
    float ope = 1.0f;
    if constexpr (FUSE) { A2row = A2 + (m_base + row16) * 128; ope = 1.0f + *epsp; }

    __builtin_prefetch(Arow, 0, 3);                 // global_prefetch_b8
    if constexpr (FUSE) __builtin_prefetch(A2row, 0, 3);

    v8f acc[NOUT / 16];
#pragma unroll
    for (int t = 0; t < NOUT / 16; ++t) acc[t] = (v8f)(0.0f);

#pragma unroll
    for (int ks = 0; ks < 4; ++ks) {
        const int k0 = ks * 32;
        const int kc = k0 + (hi << 4);              // 16 contiguous K for this half
        float4 u0 = *reinterpret_cast<const float4*>(Arow + kc);
        float4 u1 = *reinterpret_cast<const float4*>(Arow + kc + 4);
        float4 u2 = *reinterpret_cast<const float4*>(Arow + kc + 8);
        float4 u3 = *reinterpret_cast<const float4*>(Arow + kc + 12);
        if constexpr (FUSE) {
            float4 a0 = *reinterpret_cast<const float4*>(A2row + kc);
            float4 a1 = *reinterpret_cast<const float4*>(A2row + kc + 4);
            float4 a2 = *reinterpret_cast<const float4*>(A2row + kc + 8);
            float4 a3 = *reinterpret_cast<const float4*>(A2row + kc + 12);
            u0 = fuse4(u0, a0, ope); u1 = fuse4(u1, a1, ope);
            u2 = fuse4(u2, a2, ope); u3 = fuse4(u3, a3, ope);
        }
        const v16h a = pack16(u0, u1, u2, u3);

        const int kb = k0 + (hi << 4);              // B-frag: 16 contiguous K halves
#pragma unroll
        for (int nt = 0; nt < NOUT / 16; ++nt) {
            const int n = nt * 16 + row16;
            const v16h b = *reinterpret_cast<const v16h*>(&wlds[n * 128 + kb]);
            acc[nt] = __builtin_amdgcn_wmma_f32_16x16x32_f16(
                false, a, false, b, (short)0, acc[nt], false, false);
        }
    }

#pragma unroll
    for (int nt = 0; nt < NOUT / 16; ++nt) {
        const int n = nt * 16 + row16;
        float bv = 0.0f;
        if constexpr (BIAS) bv = bias[n];
#pragma unroll
        for (int r = 0; r < 8; ++r) {
            const size_t m = m_base + (size_t)r + (size_t)(hi << 3);
            float v = acc[nt][r] + bv;
            if constexpr (RELU) v = fmaxf(v, 0.0f);
            C[m * (size_t)ldc + n] = v;
        }
    }
}

// ---------------------------------------------------------------------------
// u_p[b,i,d] = sum_j (dis[b,j,i]/max(deg[b,j],1e-6)) * x[b,j,d]
// One workgroup per graph; both operands staged in LDS as f16 (64 KB total).
// ---------------------------------------------------------------------------
__global__ __launch_bounds__(256)
void up_wmma(const float* __restrict__ dis, const float* __restrict__ xbuf,
             float* __restrict__ up)
{
    __shared__ __align__(32) _Float16 a_lds[128 * 128]; // A[i][j] = dis[j,i]/deg[j]
    __shared__ __align__(32) _Float16 b_lds[128 * 128]; // B[d][j] = x[j,d] (K contiguous)
    float* dinv = reinterpret_cast<float*>(b_lds);      // 512B scratch, overwritten later

    const int tid = threadIdx.x;
    const int b   = blockIdx.x;
    const float* D = dis  + (size_t)b * 128 * 128;
    const float* X = xbuf + (size_t)b * 128 * 128;

    if (tid < 128) {
        float s = 0.0f;
        const float4* rp = reinterpret_cast<const float4*>(D + (size_t)tid * 128);
#pragma unroll 8
        for (int i = 0; i < 32; ++i) { float4 v = rp[i]; s += v.x + v.y + v.z + v.w; }
        dinv[tid] = 1.0f / fmaxf(s, 1e-6f);
    }
    __syncthreads();

    for (int idx = tid; idx < 128 * 128 / 4; idx += 256) {
        const int base = idx * 4;
        const int j = base >> 7, i0 = base & 127;
        float4 v = *reinterpret_cast<const float4*>(D + base);
        const float di = dinv[j];
        a_lds[(i0 + 0) * 128 + j] = (_Float16)(v.x * di);
        a_lds[(i0 + 1) * 128 + j] = (_Float16)(v.y * di);
        a_lds[(i0 + 2) * 128 + j] = (_Float16)(v.z * di);
        a_lds[(i0 + 3) * 128 + j] = (_Float16)(v.w * di);
    }
    __syncthreads();
    for (int idx = tid; idx < 128 * 128 / 4; idx += 256) {
        const int base = idx * 4;
        const int j = base >> 7, d0 = base & 127;
        float4 v = *reinterpret_cast<const float4*>(X + base);
        b_lds[(d0 + 0) * 128 + j] = (_Float16)v.x;
        b_lds[(d0 + 1) * 128 + j] = (_Float16)v.y;
        b_lds[(d0 + 2) * 128 + j] = (_Float16)v.z;
        b_lds[(d0 + 3) * 128 + j] = (_Float16)v.w;
    }
    __syncthreads();

    const int lane  = tid & 31;
    const int wave  = tid >> 5;
    const int row16 = lane & 15;
    const int hi    = lane >> 4;
    const int ibase = wave * 16;

    v8f acc[8];
#pragma unroll
    for (int t = 0; t < 8; ++t) acc[t] = (v8f)(0.0f);

#pragma unroll
    for (int ks = 0; ks < 4; ++ks) {
        const int k0 = ks * 32;
        const int kc = k0 + (hi << 4);
        const v8h* ap = reinterpret_cast<const v8h*>(&a_lds[(ibase + row16) * 128 + kc]);
        const v8h alo = ap[0];
        const v8h ahi = ap[1];   // +8 halves
        v16h a;
#pragma unroll
        for (int i = 0; i < 8; ++i) { a[i] = alo[i]; a[8 + i] = ahi[i]; }

        const int kb = k0 + (hi << 4);
#pragma unroll
        for (int nt = 0; nt < 8; ++nt) {
            const int n = nt * 16 + row16;
            const v16h bb = *reinterpret_cast<const v16h*>(&b_lds[n * 128 + kb]);
            acc[nt] = __builtin_amdgcn_wmma_f32_16x16x32_f16(
                false, a, false, bb, (short)0, acc[nt], false, false);
        }
    }

#pragma unroll
    for (int nt = 0; nt < 8; ++nt) {
        const int d = nt * 16 + row16;
#pragma unroll
        for (int r = 0; r < 8; ++r) {
            const int i = ibase + r + (hi << 3);
            up[((size_t)b * 128 + i) * 128 + d] = acc[nt][r];
        }
    }
}

// ---------------------------------------------------------------------------
// h[i,:] = sum_j node_emb[x_tokens[i,j], :]
// ---------------------------------------------------------------------------
__global__ void embed_nodes(const float* __restrict__ emb, const int* __restrict__ toks,
                            float* __restrict__ out)
{
    const int id = blockIdx.x * 256 + threadIdx.x;      // N*32 threads
    const int node = id >> 5;
    const int q = (id & 31) * 4;
    float4 s = make_float4(0.f, 0.f, 0.f, 0.f);
#pragma unroll
    for (int j = 0; j < 9; ++j) {
        const int t = toks[node * 9 + j];
        const float4 v = *reinterpret_cast<const float4*>(emb + (size_t)t * 128 + q);
        s.x += v.x; s.y += v.y; s.z += v.z; s.w += v.w;
    }
    *reinterpret_cast<float4*>(out + (size_t)node * 128 + q) = s;
}

// ---------------------------------------------------------------------------
// msg = relu(h[src] + sum_j edge_emb[edge_tokens[e,j]]); atomic scatter to agg[dst]
// ---------------------------------------------------------------------------
__global__ void gin_message(const float* __restrict__ h, const float* __restrict__ eemb,
                            const int* __restrict__ etoks, const int* __restrict__ eidx,
                            float* __restrict__ agg)
{
    const int id = blockIdx.x * 256 + threadIdx.x;      // E*32 threads
    const int e = id >> 5;
    const int q = (id & 31) * 4;
    const int s = eidx[e];
    const int d = eidx[E_EDGES + e];
    float4 m = *reinterpret_cast<const float4*>(h + (size_t)s * 128 + q);
#pragma unroll
    for (int j = 0; j < 3; ++j) {
        const int t = etoks[e * 3 + j];
        const float4 v = *reinterpret_cast<const float4*>(eemb + (size_t)t * 128 + q);
        m.x += v.x; m.y += v.y; m.z += v.z; m.w += v.w;
    }
    m.x = fmaxf(m.x, 0.f); m.y = fmaxf(m.y, 0.f); m.z = fmaxf(m.z, 0.f); m.w = fmaxf(m.w, 0.f);
    float* dp = agg + (size_t)d * 128 + q;
    atomicAdd(dp + 0, m.x); atomicAdd(dp + 1, m.y);
    atomicAdd(dp + 2, m.z); atomicAdd(dp + 3, m.w);
}

// ---------------------------------------------------------------------------
// batchnorm stats: stat[h]=sum, stat[128+h]=sumsq over all N rows
// ---------------------------------------------------------------------------
__global__ __launch_bounds__(256)
void bn_stats(const float* __restrict__ Z, float* __restrict__ stat)
{
    const int h = threadIdx.x & 127;
    const int part = threadIdx.x >> 7;
    const size_t row0 = (size_t)blockIdx.x * 128 + (size_t)part * 64;
    float s = 0.f, s2 = 0.f;
    for (int i = 0; i < 64; ++i) {
        const float v = Z[(row0 + i) * 128 + h];
        s += v; s2 += v * v;
    }
    atomicAdd(&stat[h], s);
    atomicAdd(&stat[128 + h], s2);
}

__global__ void bn_apply(const float* __restrict__ Z, const float* __restrict__ stat,
                         const float* __restrict__ gamma, const float* __restrict__ beta,
                         float* __restrict__ hbuf)
{
    const int id = blockIdx.x * 256 + threadIdx.x;      // N*H
    const int h = id & 127;
    const float mu  = stat[h] * (1.0f / N_NODES);
    const float var = stat[128 + h] * (1.0f / N_NODES) - mu * mu;
    const float z = (Z[id] - mu) * rsqrtf(var + 1e-5f) * gamma[h] + beta[h];
    hbuf[id] = hbuf[id] + fmaxf(z, 0.0f);
}

// g[b,h] = sum_t h[b*128+t, h]
__global__ void graph_pool(const float* __restrict__ h, float* __restrict__ g)
{
    const int id = blockIdx.x * 256 + threadIdx.x;      // G*H
    const int b = id >> 7;
    const int c = id & 127;
    float s = 0.f;
    for (int t = 0; t < 128; ++t) s += h[((size_t)b * 128 + t) * 128 + c];
    g[id] = s;
}

// causal depthwise conv (DC=4) over L then silu; reads x slice of (M,256) buffer
__global__ void conv_silu(const float* __restrict__ xr, const float* __restrict__ cw,
                          const float* __restrict__ cb, float* __restrict__ xo)
{
    const int id = blockIdx.x * 256 + threadIdx.x;      // N*H
    const int hh = id & 127;
    const int t  = (id >> 7) & 127;
    const int g  = id >> 14;
    float acc = cb[hh];
#pragma unroll
    for (int k = 0; k < DC_DIM; ++k) {
        const int tt = t + k - (DC_DIM - 1);
        if (tt >= 0)
            acc += cw[hh * DC_DIM + k] * xr[((size_t)g * 128 + tt) * 256 + hh];
    }
    xo[id] = siluf(acc);
}

// x_dbl[m, 0..11] = x[m,:] . x_proj_w[n,:]  (wave per row, shuffle reduce)
__global__ __launch_bounds__(256)
void xproj(const float* __restrict__ x, const float* __restrict__ w,
           float* __restrict__ dbl)
{
    const int wave = threadIdx.x >> 5;
    const int lane = threadIdx.x & 31;
    const size_t m = (size_t)blockIdx.x * 8 + wave;
    const float4 xv = *reinterpret_cast<const float4*>(x + m * 128 + lane * 4);
#pragma unroll
    for (int n = 0; n < XDBL_DIM; ++n) {
        const float4 wv = *reinterpret_cast<const float4*>(w + n * 128 + lane * 4);
        float p = xv.x * wv.x + xv.y * wv.y + xv.z * wv.z + xv.w * wv.w;
#pragma unroll
        for (int off = 16; off > 0; off >>= 1) p += __shfl_down(p, off, 32);
        if (lane == 0) dbl[m * XDBL_DIM + n] = p;
    }
}

// delta = softplus(x_dbl[:, :8] @ dt_w^T + dt_b)
__global__ void dt_softplus(const float* __restrict__ dbl, const float* __restrict__ dw,
                            const float* __restrict__ db, float* __restrict__ delta)
{
    const int id = blockIdx.x * 256 + threadIdx.x;      // N*H
    const int hh = id & 127;
    const size_t m = (size_t)(id >> 7);
    float s = db[hh];
#pragma unroll
    for (int k = 0; k < DT_DIM; ++k) s += dbl[m * XDBL_DIM + k] * dw[hh * DT_DIM + k];
    delta[id] = (s > 20.0f) ? s : log1pf(expf(s));
}

// selective-scan (NS=2), fused with D-skip, silu(res) gate and sum over L
__global__ void ssm_scan(const float* __restrict__ delta, const float* __restrict__ up,
                         const float* __restrict__ xv, const float* __restrict__ xr,
                         const float* __restrict__ dbl, const float* __restrict__ Alog,
                         const float* __restrict__ Dp, float* __restrict__ ysum)
{
    const int id = blockIdx.x * 256 + threadIdx.x;      // G*H
    const int hh = id & 127;
    const int g  = id >> 7;
    const float A0 = -expf(Alog[hh * 2 + 0]);
    const float A1 = -expf(Alog[hh * 2 + 1]);
    const float Dv = Dp[hh];
    float c0 = 0.f, c1 = 0.f, acc = 0.f;
    for (int t = 0; t < 128; ++t) {
        const size_t m = (size_t)g * 128 + t;
        const float dl = delta[m * 128 + hh];
        const float u  = up[m * 128 + hh];
        // B0,B1,C0,C1 live at dbl[m*12 + 8..11]; byte offset 48m+32 is 16B aligned
        const float4 bc = *reinterpret_cast<const float4*>(dbl + m * XDBL_DIM + 8);
        c0 = expf(dl * A0) * c0 + dl * bc.x * u;
        c1 = expf(dl * A1) * c1 + dl * bc.y * u;
        float y = c0 * bc.z + c1 * bc.w + xv[m * 128 + hh] * Dv;
        const float r = xr[m * 256 + 128 + hh];
        y *= siluf(r);
        acc += y;
    }
    ysum[id] = acc;
}

// out[g,n] = g_pool[g,n] + ysum[g,:] . out_proj_w[n,:]
__global__ void final_out(const float* __restrict__ ysum, const float* __restrict__ gb,
                          const float* __restrict__ ow, float* __restrict__ out)
{
    const int id = blockIdx.x * 256 + threadIdx.x;      // G*H
    const int n = id & 127;
    const int g = id >> 7;
    const float4* yv = reinterpret_cast<const float4*>(ysum + (size_t)g * 128);
    const float4* wv = reinterpret_cast<const float4*>(ow + (size_t)n * 128);
    float s = 0.f;
#pragma unroll 8
    for (int k = 0; k < 32; ++k) {
        const float4 a = yv[k], b = wv[k];
        s += a.x * b.x + a.y * b.y + a.z * b.z + a.w * b.w;
    }
    float v = gb[id] + s;
    out[id] = isfinite(v) ? v : 0.0f;
}

// ---------------------------------------------------------------------------
extern "C" void kernel_launch(void* const* d_in, const int* in_sizes, int n_in,
                              void* d_out, int out_size, void* d_ws, size_t ws_size,
                              hipStream_t stream)
{
    (void)in_sizes; (void)n_in; (void)out_size; (void)ws_size;

    const float* node_emb   = (const float*)d_in[0];
    const float* edge_emb   = (const float*)d_in[1];
    const float* eps        = (const float*)d_in[2];
    const float* mlp_w1     = (const float*)d_in[3];
    const float* mlp_b1     = (const float*)d_in[4];
    const float* mlp_w2     = (const float*)d_in[5];
    const float* mlp_b2     = (const float*)d_in[6];
    const float* bn_gamma   = (const float*)d_in[7];
    const float* bn_beta    = (const float*)d_in[8];
    const float* in_proj_w  = (const float*)d_in[9];
    const float* conv_w     = (const float*)d_in[10];
    const float* conv_b     = (const float*)d_in[11];
    const float* x_proj_w   = (const float*)d_in[12];
    const float* dt_proj_w  = (const float*)d_in[13];
    const float* dt_proj_b  = (const float*)d_in[14];
    const float* A_log      = (const float*)d_in[15];
    const float* D_param    = (const float*)d_in[16];
    const float* out_proj_w = (const float*)d_in[17];
    const float* dis_dense  = (const float*)d_in[18];
    const int*   x_tokens   = (const int*)d_in[19];
    const int*   edge_tokens= (const int*)d_in[20];
    const int*   edge_index = (const int*)d_in[21];
    /* d_in[22] (batch) is implicit: node // 128 */

    const size_t M = N_NODES;
    float* P0  = (float*)d_ws;              // h (GNN) -> u_p (Mamba)       M*128
    float* XR  = P0  + M * 128;             // GNN scratch -> in_proj out   M*256
    float* P1  = XR  + M * 256;             // x_ (conv out)  M*128
    float* P2  = P1  + M * 128;             // delta                        M*128
    float* DBL = P2  + M * 128;             // x_dbl                        M*12
    float* GB  = DBL + M * XDBL_DIM;        // graph pool                   G*128
    float* YS  = GB  + (size_t)G_DIM * 128; // ysum                         G*128
    float* ST  = YS  + (size_t)G_DIM * 128; // bn stats                     256
    float* PA  = XR;                        // agg / z2 during GNN
    float* PB  = XR + M * 128;              // z1 during GNN

    // --- embeddings ---
    embed_nodes<<<N_NODES * 32 / 256, 256, 0, stream>>>(node_emb, x_tokens, P0);

    // --- 4 GIN layers ---
    for (int li = 0; li < NL_LAYERS; ++li) {
        (void)hipMemsetAsync(PA, 0, M * 128 * sizeof(float), stream);
        gin_message<<<E_EDGES * 32 / 256, 256, 0, stream>>>(
            P0, edge_emb, edge_tokens, edge_index, PA);
        gemm_k128_wmma<128, true, true, true><<<M / 128, 256, 0, stream>>>(
            P0, PA, eps + li,
            mlp_w1 + (size_t)li * 128 * 128, mlp_b1 + li * 128, PB, 128);
        gemm_k128_wmma<128, false, false, true><<<M / 128, 256, 0, stream>>>(
            PB, nullptr, nullptr,
            mlp_w2 + (size_t)li * 128 * 128, mlp_b2 + li * 128, PA, 128);
        (void)hipMemsetAsync(ST, 0, 256 * sizeof(float), stream);
        bn_stats<<<512, 256, 0, stream>>>(PA, ST);
        bn_apply<<<M * 128 / 256, 256, 0, stream>>>(
            PA, ST, bn_gamma + li * 128, bn_beta + li * 128, P0);
    }

    // --- graph readout + Mamba block ---
    graph_pool<<<G_DIM * 128 / 256, 256, 0, stream>>>(P0, GB);
    gemm_k128_wmma<256, false, false, false><<<M / 128, 256, 0, stream>>>(
        P0, nullptr, nullptr, in_proj_w, nullptr, XR, 256);
    conv_silu<<<M * 128 / 256, 256, 0, stream>>>(XR, conv_w, conv_b, P1);
    xproj<<<M / 8, 256, 0, stream>>>(P1, x_proj_w, DBL);
    dt_softplus<<<M * 128 / 256, 256, 0, stream>>>(DBL, dt_proj_w, dt_proj_b, P2);
    up_wmma<<<G_DIM, 256, 0, stream>>>(dis_dense, P1, P0);   // h no longer needed
    ssm_scan<<<G_DIM * 128 / 256, 256, 0, stream>>>(
        P2, P0, P1, XR, DBL, A_log, D_param, YS);
    final_out<<<G_DIM * 128 / 256, 256, 0, stream>>>(YS, GB, out_proj_w, (float*)d_out);
}
